// CorticalMap_7370163880472
// MI455X (gfx1250) — compile-verified, matrix operand order checked by reading, not run on hardware
//
#include <hip/hip_runtime.h>
#include <hip/hip_bf16.h>
#include <math.h>

// ---------------------------------------------------------------------------
// CorticalMap forward for MI455X (gfx1250, wave32).
//
// Lateral 625-dim locally-connected dots -> v_wmma_f32_16x16x32_f16
// (16 locations/wave, diagonal extract), weights pre-folded (lri * 1/sum)
// into f16 B-fragments once (32 MB, L2-resident).  The K dimension is
// PERMUTED (applied to both A and B, dot-invariant) into runs of 8
// consecutive kj so that, with the chunk loop fully unrolled, every A-gather
// is a ds_load_u16 with a compile-time immediate offset from one per-lane
// base address (no runtime /25, %25).  Activity map staged in LDS as two
// column-parity planes of the 25 even rows actually touched (10.4 KB).
// ---------------------------------------------------------------------------

typedef __attribute__((ext_vector_type(16))) _Float16 v16h;
typedef __attribute__((ext_vector_type(8)))  float    v8f;

#define SHEET 160
#define NLOC  (SHEET*SHEET)          // 25600
#define AFF_K 15
#define F_AFF 450                    // 2*15*15
#define IN_HW 174                    // SHEET + AFF_K - 1
#define LAT_K 25
#define F_LAT 625
#define NCHUNK 20                    // 640 / 32 (K padded)
#define NGROUP (NLOC/16)             // 1600 groups of 16 locations
#define GPR 10                       // groups per sheet row
#define PLANE 2600                   // 25 rows * 104 cols per parity plane
#define TILE_ELEMS (2*PLANE)         // 5200 halves
#define HOMEO 0.02f
#define BAL 0.55f
#define LAT_ITERS 10

// workspace layout (float indices; wB region byte-addressed, 32B aligned)
#define OFF_SRE   0
#define OFF_LRI   64
#define OFF_ENV   704
#define OFF_ISUM  960
#define OFF_AFF   26560
#define OFF_LAT   52160
#define OFF_LATM  77760
#define OFF_CONV  103360
#define OFF_PART  128960
#define OFF_WB_BYTES (129152u*4u)    // then 1600*20*512 halves = 32.768 MB

// ISA 16-bit A-matrix 16x32 half->K map (B assumed symmetric, N = lane&15):
//   lanes 0-15 : h<8 -> K=h,   h>=8 -> K=h+8
//   lanes 16-31: h<8 -> K=h+8, h>=8 -> K=h+16
__device__ __forceinline__ int kmap(int lane, int h) {
  return ((h < 8) ? h : h + 8) + ((lane & 16) ? 8 : 0);
}

// K-slot permutation: slot s (0..639), run = s/8, t = s%8.
//   run < 75 : feature f = ki*25 + kjb*8 + t   (ki = run/3, kjb = run%3)
//   run >= 75: e = (run-75)*8 + t; e<25 -> f = e*25+24 (kj=24 col); else pad.
// LDS element offset (within a parity plane, relative to lane base c>>1):
//   f=(ki,kj) -> ki*104 + kj
constexpr int offs_of_slot(int s) {
  int run = s >> 3, t = s & 7;
  if (run < 75) { int ki = run/3, kjb = run - ki*3; return ki*104 + kjb*8 + t; }
  int e = (run - 75)*8 + t;
  if (e > 24) e = 24;               // pad slots: weight is 0, any valid addr
  return e*104 + 24;
}

// -------------------------------- constants --------------------------------
__global__ void k_init(float* __restrict__ ws) {
  if (threadIdx.x != 0 || blockIdx.x != 0) return;
  float* sre = ws + OFF_SRE;
  float* lri = ws + OFF_LRI;
  float* env = ws + OFF_ENV;
  const float HPI = 1.57079632679f;
  float mx = 0.f;
  for (int i = 0; i < 225; ++i) {                 // afferent envelope 15x15
    float cc = 7.f, dy = (float)(i/15) - cc, dx = (float)(i%15) - cc;
    float d = sqrtf(dy*dy + dx*dx);
    float v = cosf(fminf(d/15.f, 1.f)*HPI); v *= v;
    v *= (d < 7.5f) ? 1.f : 0.f;
    env[i] = v; mx = fmaxf(mx, v);
  }
  for (int i = 0; i < 225; ++i) env[i] /= mx;
  float s = 0.f;
  for (int i = 0; i < 25; ++i) {                  // sre 5x5 (/sum)
    float cc = 2.f, dy = (float)(i/5) - cc, dx = (float)(i%5) - cc;
    float d = sqrtf(dy*dy + dx*dx);
    float v = cosf(fminf(d/5.f, 1.f)*HPI); v *= v;
    v *= (d < 2.5f) ? 1.f : 0.f;
    sre[i] = v; s += v;
  }
  for (int i = 0; i < 25; ++i) sre[i] /= s;
  mx = 0.f;
  for (int i = 0; i < 625; ++i) {                 // lri 25x25 (/max)
    float cc = 12.f, dy = (float)(i/25) - cc, dx = (float)(i%25) - cc;
    float d = sqrtf(dy*dy + dx*dx);
    float inh = cosf(fminf(d/2.5f, 1.f)*HPI); inh *= inh;
    inh *= (d < 1.25f) ? 1.f : 0.f;
    float v = cosf(fminf(d/25.f, 1.f)*HPI); v *= v;
    v *= (d < 12.5f) ? 1.f : 0.f;
    v *= (1.f - inh);
    lri[i] = v; mx = fmaxf(mx, v);
  }
  for (int i = 0; i < 625; ++i) lri[i] /= mx;
  for (int i = 625; i < 640; ++i) lri[i] = 0.f;
}

// -------- per-location lateral weight sums (for 1/sum normalization) -------
__global__ void k_wsum(const float* __restrict__ latw, float* __restrict__ inv_sum) {
  int wid  = (blockIdx.x*blockDim.x + threadIdx.x) >> 5;
  int lane = threadIdx.x & 31;
  if (wid >= NLOC) return;
  const float* wp = latw + (size_t)wid * F_LAT;
  float s = 0.f;
  for (int f = lane; f < F_LAT; f += 32) s += wp[f];
  for (int o = 16; o > 0; o >>= 1) s += __shfl_xor(s, o, 32);
  if (lane == 0) inv_sum[wid] = 1.f / s;
}

// ---- pack folded lateral weights (latw*lri/sum), permuted, B fragments ----
__global__ void k_wfrag(const float* __restrict__ latw, const float* __restrict__ lri,
                        const float* __restrict__ inv_sum, _Float16* __restrict__ wB) {
  int wid  = (blockIdx.x*blockDim.x + threadIdx.x) >> 5;   // group*NCHUNK + chunk
  int lane = threadIdx.x & 31;
  if (wid >= NGROUP*NCHUNK) return;
  int g = wid / NCHUNK, k = wid - g*NCHUNK;
  int loc = g*16 + (lane & 15);
  float inv = inv_sum[loc];
  const float* wp = latw + (size_t)loc * F_LAT;
  v16h b;
#pragma unroll
  for (int h = 0; h < 16; ++h) {
    int s = 32*k + kmap(lane, h);
    int run = s >> 3, t = s & 7;
    float v = 0.f;
    if (run < 75) {
      int ki = run/3, kjb = run - ki*3;
      int f = ki*25 + kjb*8 + t;
      v = wp[f]*lri[f]*inv;
    } else {
      int e = (run - 75)*8 + t;
      if (e < 25) { int f = e*25 + 24; v = wp[f]*lri[f]*inv; }
    }
    b[h] = (_Float16)v;
  }
  *(v16h*)(wB + (size_t)wid*512 + lane*16) = b;            // 32B/lane, coalesced
}

// ------------- afferent locally-connected pass (one wave/location) ---------
__global__ void k_aff(const float* __restrict__ x, const float* __restrict__ rfs,
                      const float* __restrict__ adath, const float* __restrict__ env,
                      float* __restrict__ out_raw, float* __restrict__ aff,
                      float* __restrict__ lat, float* __restrict__ latm) {
  int wid  = (blockIdx.x*blockDim.x + threadIdx.x) >> 5;
  int lane = threadIdx.x & 31;
  if (wid >= NLOC) return;
  int r = wid / SHEET, c = wid - r*SHEET;
  const float* rp = rfs + (size_t)wid * F_AFF;
  float dot = 0.f, s = 0.f;
  for (int f = lane; f < F_AFF; f += 32) {
    int ch = f / 225, rem = f - ch*225;
    int i = rem / 15, j = rem - i*15;
    float xv = x[ch*IN_HW*IN_HW + (r+i)*IN_HW + (c+j)];
    float wv = rp[f];
    dot += xv * env[rem] * wv;
    s   += wv;
  }
  for (int o = 16; o > 0; o >>= 1) { dot += __shfl_xor(dot, o, 32); s += __shfl_xor(s, o, 32); }
  if (lane == 0) {
    float raw = dot * 45.f / s;
    out_raw[wid] = raw;
    float a = raw / 45.f - adath[wid];
    aff[wid]  = a;
    lat[wid]  = fmaxf(a, 0.f);
    latm[wid] = 0.f;
  }
}

// ---------------------- 5x5 sre convolution (pad 0.02) ---------------------
__global__ void k_conv(const float* __restrict__ lat, const float* __restrict__ sre,
                       float* __restrict__ conv) {
  int loc = blockIdx.x*blockDim.x + threadIdx.x;
  if (loc >= NLOC) return;
  int r = loc / SHEET, c = loc - r*SHEET;
  float s = 0.f;
#pragma unroll
  for (int i = 0; i < 5; ++i)
#pragma unroll
    for (int j = 0; j < 5; ++j) {
      int rr = r + i - 2, cc = c + j - 2;
      float v = (rr >= 0 && rr < SHEET && cc >= 0 && cc < SHEET) ? lat[rr*SHEET+cc] : HOMEO;
      s = fmaf(sre[i*5+j], v, s);
    }
  conv[loc] = s;
}

// -------- lateral WMMA step: one block per sheet row, 10 waves/row ---------
// mode 0: lat update (lat_neg from conv map)   mode 1: hebbian (from lat_mean/10)
__global__ __launch_bounds__(320)
void k_lat_step(const float* __restrict__ src, const float* __restrict__ aff,
                float* __restrict__ lat, float* __restrict__ latm,
                const _Float16* __restrict__ wB,
                float* __restrict__ out_lat, float* __restrict__ partial,
                int mode, int writeOut) {
  __shared__ _Float16 tileP[TILE_ELEMS];  // [parity][25 even rows][104 cols]
  __shared__ float    diag[SHEET];
  const int r = blockIdx.x, tid = threadIdx.x;
  const float srcScale = (mode == 1) ? 0.1f : 1.0f;

  // stage pad24(src*scale, 0.02): only the 25 even row-offsets are touched
  for (int i = tid; i < TILE_ELEMS; i += 320) {
    int p = i / PLANE, rem = i - p*PLANE;
    int row = rem / 104, col2 = rem - row*104;
    int rr = r + 2*row - 24;
    int cc = 2*col2 + p - 24;
    float v = (rr >= 0 && rr < SHEET && cc >= 0 && cc < SHEET) ? src[rr*SHEET+cc]*srcScale
                                                               : HOMEO;
    tileP[i] = (_Float16)v;
  }
  __syncthreads();

  const int g = tid >> 5;                   // group (wave) 0..9 -> cols 16g..
  const int lane = tid & 31;
  const bool upper = (lane & 16) != 0;
  const int c = g*16 + (lane & 15);         // this lane's sheet column (A row M)
  const _Float16* lanebase = tileP + (c & 1)*PLANE + (c >> 1);
  const size_t gbase = ((size_t)(r*GPR + g) * NCHUNK) * 512;

  v8f acc = {};
#pragma unroll
  for (int k = 0; k < NCHUNK; ++k) {
    v16h b = *(const v16h*)(wB + gbase + (size_t)k*512 + lane*16);
    v16h a;
    if (k < 18) {
      // runs 4k..4k+3 are all "full" runs: base select + immediate offsets
      const _Float16* p0 = lanebase + (upper ? offs_of_slot((4*k+1)*8)
                                             : offs_of_slot((4*k  )*8));
      const _Float16* p1 = lanebase + (upper ? offs_of_slot((4*k+3)*8)
                                             : offs_of_slot((4*k+2)*8));
#pragma unroll
      for (int t = 0; t < 8; ++t) { a[t] = p0[t]; a[t+8] = p1[t]; }
    } else {
      // tail chunks include remainder runs (kj==24 column / pads)
#pragma unroll
      for (int h = 0; h < 16; ++h) {
        const int s0 = 32*k + ((h < 8) ? h : h + 8);
        a[h] = lanebase[upper ? offs_of_slot(s0 + 8) : offs_of_slot(s0)];
      }
    }
    acc = __builtin_amdgcn_wmma_f32_16x16x32_f16(false, a, false, b,
                                                 (short)0, acc, false, false);
  }
  // diagonal of 16x16 f32 C: (i,i) i<8 -> lane i, vgpr i; i>=8 -> lane i+16, vgpr i-8
  float d = 0.f;
#pragma unroll
  for (int v = 0; v < 8; ++v) {
    d = (lane == v)      ? acc[v] : d;
    d = (lane == v + 24) ? acc[v] : d;
  }
  if (lane < 8)        diag[g*16 + lane]      = d;
  else if (lane >= 24) diag[g*16 + lane - 16] = d;
  __syncthreads();

  if (mode == 0) {
    if (tid < SHEET) {
      int loc = r*SHEET + tid;
      float o = src[loc]*(1.f - BAL) - diag[tid]*BAL;  // src == conv map
      o = o + aff[loc];                                // STRENGTH = 1
      o = tanhf(fmaxf(o, 0.f));
      lat[loc]   = o;
      latm[loc] += o;
      if (writeOut) out_lat[loc] = o;
    }
  } else {
    float contrib = 0.f;
    if (tid < SHEET) {
      int loc = r*SHEET + tid;
      contrib = 62.5f * (src[loc]*0.1f) * diag[tid];   // latw_n = 62.5 * folded w
    }
    __syncthreads();
    if (tid < SHEET) diag[tid] = contrib;
    __syncthreads();
    if (tid == 0) {                                    // fixed-order row partial
      float s = 0.f;
      for (int i = 0; i < SHEET; ++i) s += diag[i];
      partial[r] = s;
    }
  }
}

// ----------------- deterministic final scalar reduction --------------------
__global__ void k_final(const float* __restrict__ partial, float* __restrict__ out_scalar) {
  if (threadIdx.x == 0 && blockIdx.x == 0) {
    float s = 0.f;
    for (int i = 0; i < SHEET; ++i) s += partial[i];
    *out_scalar = s;
  }
}

extern "C" void kernel_launch(void* const* d_in, const int* in_sizes, int n_in,
                              void* d_out, int out_size, void* d_ws, size_t ws_size,
                              hipStream_t stream) {
  (void)in_sizes; (void)n_in; (void)out_size; (void)ws_size; // needs ~33.3 MB ws
  const float* x    = (const float*)d_in[0];   // (1,2,174,174)
  const float* rfs  = (const float*)d_in[1];   // (25600,450,1)
  const float* latw = (const float*)d_in[2];   // (25600,625,1)
  const float* adat = (const float*)d_in[3];   // (1,1,160,160) zeros
  float* out = (float*)d_out;                  // [raw_aff | lat | corr] = 51201
  float* ws  = (float*)d_ws;

  float* sre  = ws + OFF_SRE;
  float* lri  = ws + OFF_LRI;
  float* env  = ws + OFF_ENV;
  float* isum = ws + OFF_ISUM;
  float* aff  = ws + OFF_AFF;
  float* lat  = ws + OFF_LAT;
  float* latm = ws + OFF_LATM;
  float* conv = ws + OFF_CONV;
  float* part = ws + OFF_PART;
  _Float16* wB = (_Float16*)((char*)d_ws + OFF_WB_BYTES);

  k_init <<<1, 1, 0, stream>>>(ws);
  k_wsum <<<(NLOC*32)/256, 256, 0, stream>>>(latw, isum);
  k_wfrag<<<(NGROUP*NCHUNK*32)/256, 256, 0, stream>>>(latw, lri, isum, wB);
  k_aff  <<<(NLOC*32)/256, 256, 0, stream>>>(x, rfs, adat, env, out, aff, lat, latm);

  for (int it = 0; it < LAT_ITERS; ++it) {
    k_conv    <<<NLOC/256, 256, 0, stream>>>(lat, sre, conv);
    k_lat_step<<<SHEET, 320, 0, stream>>>(conv, aff, lat, latm, wB,
                                          out + NLOC, part, 0, (it == LAT_ITERS-1) ? 1 : 0);
  }
  // hebbian correlation reuses the same WMMA path on lat_mean/10
  k_lat_step<<<SHEET, 320, 0, stream>>>(latm, aff, lat, latm, wB,
                                        out + NLOC, part, 1, 0);
  k_final  <<<1, 1, 0, stream>>>(part, out + 2*NLOC);
}
